// NA2QMixer_8598524526933
// MI455X (gfx1250) — matrix operand order repaired and checked by reading, not compiled, not run on hardware
//
#include <hip/hip_runtime.h>
#include <hip/hip_bf16.h>
#include <math.h>

#define B_TOTAL 32768
#define NA 32
#define SD 512
#define H 64
#define N1 32
#define N2 496
#define NF 528
#define MT 32
#define NTHREADS 256
#define NWAVES 8

// f16 transposed weights in d_ws (element offsets)
#define WS_T_OFF   0
#define WZ_T_OFF   32768
#define WB1_T_OFF  65536
#define WA1_T_OFF  98304
#define WA2_T_OFF  106496
#define WT_TOTAL   140288

// LDS layout (bytes), all 16B aligned
#define SA_STRIDE 520                 // f16 elems/row (512 + pad)
#define OFF_AST 0
#define SZ_AST (MT*SA_STRIDE*2)       // 33280
#define OFF_ASE (OFF_AST + SZ_AST)    // 33280
#define OFF_Q   (OFF_ASE + SZ_AST)    // 66560
#define Q_STRIDE 33
#define SZ_Q (MT*Q_STRIDE*4)          // 4224
#define OFF_COMB (OFF_Q + SZ_Q)       // 70784
#define COMB_STRIDE 136
#define SZ_COMB (MT*COMB_STRIDE*2)    // 8704
#define OFF_HA (OFF_COMB + SZ_COMB)   // 79488
#define HA_STRIDE 72
#define SZ_HA (MT*HA_STRIDE*2)        // 4608
#define OFF_BH (OFF_HA + SZ_HA)       // 84096
#define BH_STRIDE 68
#define SZ_BH (MT*BH_STRIDE*4)        // 8704
#define OFF_LOG (OFF_BH + SZ_BH)      // 92800
#define LG_STRIDE 529
#define SZ_LOG (MT*LG_STRIDE*4)       // 67712
#define OFF_SHP (OFF_LOG + SZ_LOG)    // 160512
#define SZ_SHP (MT*LG_STRIDE*4)       // 67712
#define SMEM_BYTES (OFF_SHP + SZ_SHP) // 228224 (< 320KB/WGP)

typedef __attribute__((ext_vector_type(16))) _Float16 v16h;
typedef __attribute__((ext_vector_type(8)))  float    v8f;

union Frag { v16h v; uint4 u[2]; };

// A fragment (16x32 f16) from LDS; lane<16: K {k0..k0+7, k0+16..k0+23}, lane>=16: +8
__device__ __forceinline__ v16h lda16(const _Float16* tileRow0, int stride, int k0, int lane) {
  int m = lane & 15, h = lane >> 4;
  const _Float16* p = tileRow0 + m * stride + k0 + h * 8;
  Frag f;
  f.u[0] = *(const uint4*)p;
  f.u[1] = *(const uint4*)(p + 16);
  return f.v;
}

// B fragment (32x16 f16) from L2-resident transposed weights [N][K]
__device__ __forceinline__ v16h ldb16(const _Float16* wT, int Kdim, int n0, int k0, int lane) {
  int n = n0 + (lane & 15), h = lane >> 4;
  const uint4* p = (const uint4*)(wT + (size_t)n * Kdim + k0 + h * 8);
  Frag f;
  f.u[0] = p[0];
  f.u[1] = p[2];
  return f.v;
}

// ---- prep: convert + transpose weights to f16 [N][K] in workspace ----
__global__ void na2q_prep(const float* __restrict__ ws_w, const float* __restrict__ wz_w,
                          const float* __restrict__ wb1_w, const float* __restrict__ wa1_w,
                          const float* __restrict__ wa2_w, _Float16* __restrict__ wt) {
  int i = blockIdx.x * blockDim.x + threadIdx.x;
  if (i >= WT_TOTAL) return;
  float v;
  if (i < WZ_T_OFF)       { int l = i;             int n = l >> 9, k = l & 511; v = ws_w [k * H  + n]; }
  else if (i < WB1_T_OFF) { int l = i - WZ_T_OFF;  int n = l >> 9, k = l & 511; v = wz_w [k * H  + n]; }
  else if (i < WA1_T_OFF) { int l = i - WB1_T_OFF; int n = l >> 9, k = l & 511; v = wb1_w[k * H  + n]; }
  else if (i < WA2_T_OFF) { int l = i - WA1_T_OFF; int n = l >> 7, k = l & 127; v = wa1_w[k * H  + n]; }
  else                    { int l = i - WA2_T_OFF; int n = l >> 6, k = l & 63;  v = wa2_w[k * NF + n]; }
  wt[i] = (_Float16)v;
}

// ---- main fused kernel: one block = 32 batch rows ----
__global__ __launch_bounds__(NTHREADS)
void na2q_main(const float* __restrict__ q, const float* __restrict__ state, const float* __restrict__ sem,
               const float* __restrict__ w1_1, const float* __restrict__ b1_1,
               const float* __restrict__ w2_1, const float* __restrict__ b2_1,
               const float* __restrict__ w3_1, const float* __restrict__ b3_1,
               const float* __restrict__ w1_2, const float* __restrict__ b1_2,
               const float* __restrict__ w2_2, const float* __restrict__ b2_2,
               const float* __restrict__ w3_2, const float* __restrict__ b3_2,
               const float* __restrict__ ws_b, const float* __restrict__ wz_b,
               const float* __restrict__ wa1_b, const float* __restrict__ wa2_b,
               const float* __restrict__ wb1_b, const float* __restrict__ wb2_w,
               const float* __restrict__ wb2_b,
               const _Float16* __restrict__ wt,
               float* __restrict__ out_q, float* __restrict__ out_att, float* __restrict__ out_shp) {
  extern __shared__ char smem[];
  _Float16* sAst  = (_Float16*)(smem + OFF_AST);
  _Float16* sAse  = (_Float16*)(smem + OFF_ASE);
  float*    sQ    = (float*)   (smem + OFF_Q);
  _Float16* sComb = (_Float16*)(smem + OFF_COMB);
  _Float16* sHa   = (_Float16*)(smem + OFF_HA);
  float*    sBH   = (float*)   (smem + OFF_BH);
  float*    sLog  = (float*)   (smem + OFF_LOG);
  float*    sShp  = (float*)   (smem + OFF_SHP);

  const int  tid  = threadIdx.x;
  const int  lane = tid & 31;
  const int  wave = tid >> 5;
  const long b0   = (long)blockIdx.x * MT;

  const _Float16* wsT  = wt + WS_T_OFF;
  const _Float16* wzT  = wt + WZ_T_OFF;
  const _Float16* wb1T = wt + WB1_T_OFF;
  const _Float16* wa1T = wt + WA1_T_OFF;
  const _Float16* wa2T = wt + WA2_T_OFF;

  // warm L2/WGP$ with the big attention-output weight block (global_prefetch_b8)
  if (wave == 0) __builtin_prefetch(wa2T + lane * 1056, 0, 1);

  // ---- stage activations to LDS (f32 -> f16), q stays f32 ----
  for (int idx = tid; idx < MT * (SD / 4); idx += NTHREADS) {
    int row = idx >> 7;
    int c4  = (idx & 127) << 2;
    float4 v = *(const float4*)(state + (b0 + row) * SD + c4);
    union { _Float16 h[4]; uint2 u; } pk;
    pk.h[0] = (_Float16)v.x; pk.h[1] = (_Float16)v.y; pk.h[2] = (_Float16)v.z; pk.h[3] = (_Float16)v.w;
    *(uint2*)(sAst + row * SA_STRIDE + c4) = pk.u;
    float4 w = *(const float4*)(sem + (b0 + row) * SD + c4);
    pk.h[0] = (_Float16)w.x; pk.h[1] = (_Float16)w.y; pk.h[2] = (_Float16)w.z; pk.h[3] = (_Float16)w.w;
    *(uint2*)(sAse + row * SA_STRIDE + c4) = pk.u;
  }
  for (int idx = tid; idx < MT * NA; idx += NTHREADS) {
    int row = idx >> 5, c = idx & 31;
    sQ[row * Q_STRIDE + c] = q[(b0 + row) * NA + c];
  }
  __syncthreads();

  // ---- Phase A: three K=512 GEMMs (state/sem/bias encoders), 24 tiles / 8 waves ----
  for (int t = 0; t < 3; ++t) {
    int task = wave * 3 + t;
    int mat = task >> 3, sub = task & 7;
    int rowtile = sub >> 2, ntile = sub & 3;
    const _Float16* A    = (mat == 1) ? sAse : sAst;
    const _Float16* Bw   = (mat == 0) ? wsT : (mat == 1) ? wzT : wb1T;
    const float*    bias = (mat == 0) ? ws_b : (mat == 1) ? wz_b : wb1_b;
    const _Float16* Arow0 = A + rowtile * 16 * SA_STRIDE;
    v8f c = {};
    for (int k0 = 0; k0 < SD; k0 += 32) {
      v16h a = lda16(Arow0, SA_STRIDE, k0, lane);
      v16h b = ldb16(Bw, SD, ntile * 16, k0, lane);
      c = __builtin_amdgcn_wmma_f32_16x16x32_f16(false, a, false, b, (short)0, c, false, false);
    }
    int ncol = ntile * 16 + (lane & 15);
    float bv = bias[ncol];
    int h8 = (lane >> 4) * 8;
    for (int r = 0; r < 8; ++r) {
      int m = rowtile * 16 + r + h8;
      float x = fmaxf(c[r] + bv, 0.0f);
      if (mat == 2) sBH[m * BH_STRIDE + ncol] = x;
      else          sComb[m * COMB_STRIDE + (mat == 1 ? H : 0) + ncol] = (_Float16)x;
    }
  }

  // ---- Shape functions: lane = sample, wave-uniform function index (scalar weight loads) ----
  for (int n = wave; n < NF; n += NWAVES) {
    const float *W1, *B1, *W2, *B2, *W3;
    float b3v, x0, x1;
    bool o2 = (n >= N1);
    if (!o2) {
      W1 = w1_1 + n * 8; B1 = b1_1 + n * 8; W2 = w2_1 + n * 32; B2 = b2_1 + n * 4; W3 = w3_1 + n * 4;
      b3v = b3_1[n];
      x0 = sQ[lane * Q_STRIDE + n]; x1 = 0.0f;
    } else {
      int p = n - N1;
      int i = 0, rem = p;
      while (rem >= (NA - 1 - i)) { rem -= (NA - 1 - i); ++i; }  // triu_indices order
      int j = i + 1 + rem;
      W1 = w1_2 + p * 16; B1 = b1_2 + p * 8; W2 = w2_2 + p * 32; B2 = b2_2 + p * 4; W3 = w3_2 + p * 4;
      b3v = b3_2[p];
      x0 = sQ[lane * Q_STRIDE + i]; x1 = sQ[lane * Q_STRIDE + j];
    }
    float hbuf[8];
    for (int k = 0; k < 8; ++k) {
      float a = fabsf(W1[k]) * x0 + B1[k];
      if (o2) a += fabsf(W1[8 + k]) * x1;
      hbuf[k] = (a > 0.0f) ? a : expm1f(a);   // ELU
    }
    float f = b3v;
    for (int k2 = 0; k2 < 4; ++k2) {
      float g = B2[k2];
      for (int k = 0; k < 8; ++k) g += hbuf[k] * fabsf(W2[k * 4 + k2]);
      g = (g > 0.0f) ? g : expm1f(g);
      f += g * fabsf(W3[k2]);
    }
    sShp[lane * LG_STRIDE + n] = f;
  }
  __syncthreads();

  // ---- Phase B: combined[32x128] @ wa1[128x64], 8 tiles ----
  {
    int rowtile = wave >> 2, ntile = wave & 3;
    const _Float16* Arow0 = sComb + rowtile * 16 * COMB_STRIDE;
    v8f c = {};
    for (int k0 = 0; k0 < 2 * H; k0 += 32) {
      v16h a = lda16(Arow0, COMB_STRIDE, k0, lane);
      v16h b = ldb16(wa1T, 2 * H, ntile * 16, k0, lane);
      c = __builtin_amdgcn_wmma_f32_16x16x32_f16(false, a, false, b, (short)0, c, false, false);
    }
    int ncol = ntile * 16 + (lane & 15);
    float bv = wa1_b[ncol];
    int h8 = (lane >> 4) * 8;
    for (int r = 0; r < 8; ++r) {
      int m = rowtile * 16 + r + h8;
      float x = fmaxf(c[r] + bv, 0.0f);
      sHa[m * HA_STRIDE + ncol] = (_Float16)x;
    }
  }
  __syncthreads();

  // ---- Phase C: ha[32x64] @ wa2[64x528] -> logits, 66 tiles ----
  for (int task = wave; task < 66; task += NWAVES) {
    int rowtile = task & 1, ntile = task >> 1;
    const _Float16* Arow0 = sHa + rowtile * 16 * HA_STRIDE;
    v8f c = {};
    for (int k0 = 0; k0 < H; k0 += 32) {
      v16h a = lda16(Arow0, HA_STRIDE, k0, lane);
      v16h b = ldb16(wa2T, H, ntile * 16, k0, lane);
      c = __builtin_amdgcn_wmma_f32_16x16x32_f16(false, a, false, b, (short)0, c, false, false);
    }
    int ncol = ntile * 16 + (lane & 15);
    float bv = wa2_b[ncol];
    int h8 = (lane >> 4) * 8;
    for (int r = 0; r < 8; ++r) {
      int m = rowtile * 16 + r + h8;
      sLog[m * LG_STRIDE + ncol] = c[r] + bv;
    }
  }
  __syncthreads();

  // ---- shape_outputs writeout (coalesced) ----
  for (int idx = tid; idx < MT * NF; idx += NTHREADS) {
    int row = idx / NF, col = idx - row * NF;
    out_shp[(b0 + row) * NF + col] = sShp[row * LG_STRIDE + col];
  }

  // ---- per-row softmax, weighted sum, bias head (wave32 shuffle reductions) ----
  for (int rr = 0; rr < MT / NWAVES; ++rr) {
    int row = wave * (MT / NWAVES) + rr;
    float* lrow = sLog + row * LG_STRIDE;
    float mx = -3.4e38f;
    for (int c = lane; c < NF; c += 32) mx = fmaxf(mx, lrow[c]);
    for (int o = 16; o > 0; o >>= 1) mx = fmaxf(mx, __shfl_xor(mx, o));
    float s = 0.0f;
    for (int c = lane; c < NF; c += 32) { float e = expf(lrow[c] - mx); lrow[c] = e; s += e; }
    for (int o = 16; o > 0; o >>= 1) s += __shfl_xor(s, o);
    float inv = 1.0f / s;
    float acc = 0.0f;
    for (int c = lane; c < NF; c += 32) {
      float aw = lrow[c] * inv;
      out_att[(b0 + row) * NF + c] = aw;
      acc += aw * sShp[row * LG_STRIDE + c];
    }
    // bias head: relu(state@wb1) . wb2 (64 terms, 2 per lane)
    acc += sBH[row * BH_STRIDE + lane]      * wb2_w[lane]
         + sBH[row * BH_STRIDE + lane + 32] * wb2_w[lane + 32];
    for (int o = 16; o > 0; o >>= 1) acc += __shfl_xor(acc, o);
    if (lane == 0) out_q[b0 + row] = acc + wb2_b[0];
  }
}

extern "C" void kernel_launch(void* const* d_in, const int* in_sizes, int n_in,
                              void* d_out, int out_size, void* d_ws, size_t ws_size,
                              hipStream_t stream) {
  (void)in_sizes; (void)n_in; (void)out_size; (void)ws_size;
  const float* q     = (const float*)d_in[0];
  const float* state = (const float*)d_in[1];
  const float* sem   = (const float*)d_in[2];
  const float* w1_1  = (const float*)d_in[3];
  const float* b1_1  = (const float*)d_in[4];
  const float* w2_1  = (const float*)d_in[5];
  const float* b2_1  = (const float*)d_in[6];
  const float* w3_1  = (const float*)d_in[7];
  const float* b3_1  = (const float*)d_in[8];
  const float* w1_2  = (const float*)d_in[9];
  const float* b1_2  = (const float*)d_in[10];
  const float* w2_2  = (const float*)d_in[11];
  const float* b2_2  = (const float*)d_in[12];
  const float* w3_2  = (const float*)d_in[13];
  const float* b3_2  = (const float*)d_in[14];
  const float* ws_w  = (const float*)d_in[15];
  const float* ws_b  = (const float*)d_in[16];
  const float* wz_w  = (const float*)d_in[17];
  const float* wz_b  = (const float*)d_in[18];
  const float* wa1_w = (const float*)d_in[19];
  const float* wa1_b = (const float*)d_in[20];
  const float* wa2_w = (const float*)d_in[21];
  const float* wa2_b = (const float*)d_in[22];
  const float* wb1_w = (const float*)d_in[23];
  const float* wb1_b = (const float*)d_in[24];
  const float* wb2_w = (const float*)d_in[25];
  const float* wb2_b = (const float*)d_in[26];

  _Float16* wt = (_Float16*)d_ws;
  float* out   = (float*)d_out;
  float* out_q   = out;
  float* out_att = out + B_TOTAL;
  float* out_shp = out + B_TOTAL + (size_t)B_TOTAL * NF;

  na2q_prep<<<WT_TOTAL / 256, 256, 0, stream>>>(ws_w, wz_w, wb1_w, wa1_w, wa2_w, wt);

  na2q_main<<<B_TOTAL / MT, NTHREADS, SMEM_BYTES, stream>>>(
      q, state, sem,
      w1_1, b1_1, w2_1, b2_1, w3_1, b3_1,
      w1_2, b1_2, w2_2, b2_2, w3_2, b3_2,
      ws_b, wz_b, wa1_b, wa2_b, wb1_b, wb2_w, wb2_b,
      wt, out_q, out_att, out_shp);
}
